// CollapseEngine_66417374265588
// MI455X (gfx1250) — matrix operand order repaired and checked by reading, not compiled, not run on hardware
//
#include <hip/hip_runtime.h>
#include <hip/hip_bf16.h>

typedef __attribute__((ext_vector_type(16))) _Float16 v16h;
typedef __attribute__((ext_vector_type(8)))  _Float16 v8h;
typedef __attribute__((ext_vector_type(8)))  float    v8f;

#define DIM    256
#define LAYERS 6
#define WSTR   264   // padded LDS row stride in halves (256 + 8) -> conflict-free b128 reads

static constexpr int    W_ELEMS    = 256 * WSTR;   // halves per staged weight matrix
static constexpr int    TILE_ELEMS = 16 * WSTR;    // halves per per-wave h/t scratch tile
static constexpr size_t SMEM_BYTES =
    (size_t)(2 * W_ELEMS + 4 * TILE_ELEMS) * sizeof(_Float16) +
    (size_t)(768 + 256 + 256 + 4) * sizeof(float);

__device__ __forceinline__ v16h cat16(v8h lo, v8h hi) {
  return __builtin_shufflevector(lo, hi, 0,1,2,3,4,5,6,7,8,9,10,11,12,13,14,15);
}

// Branch-free tanh: gfx1250 V_TANH_F32 transcendental when available; otherwise
// tanh(x) = 1 - 2/(exp2(2*log2(e)*x)+1) via v_exp_f32 + v_rcp_f32.
__device__ __forceinline__ float fast_tanh(float x) {
#if __has_builtin(__builtin_amdgcn_tanhf)
  return __builtin_amdgcn_tanhf(x);
#else
  float e = __builtin_amdgcn_exp2f(x * 2.8853900817779268f);  // 2*log2(e)
  return 1.0f - 2.0f * __builtin_amdgcn_rcpf(e + 1.0f);
#endif
}

// A-fragment (16x32 f16, M x K): lane row = l&15; half-group hg picks K-subblock.
// Element c -> K = k0 + 8*hg + (c<8 ? c : c+8): two contiguous 8-half chunks.
__device__ __forceinline__ v16h ldA(const _Float16* base, int row, int k0, int hg) {
  const _Float16* p = base + row * WSTR + k0 + hg * 8;
  v8h lo = *(const v8h*)(p);
  v8h hi = *(const v8h*)(p + 16);
  return cat16(lo, hi);
}

// B-fragment (32x16 f16, K x N): lane col n = l&15 reads weight row j (B[k][j]=W[j][k]);
// Element c -> K = k0 + 8*hg + (c<8 ? c : c+8), matching ldA.
__device__ __forceinline__ v16h ldB(const _Float16* base, int row, int k0, int hg) {
  const _Float16* p = base + row * WSTR + k0 + hg * 8;
  v8h lo = *(const v8h*)(p);
  v8h hi = *(const v8h*)(p + 16);
  return cat16(lo, hi);
}

extern "C" __global__ __launch_bounds__(128, 1)
void collapse_engine_kernel(const float* __restrict__ h0,
                            const float* __restrict__ W1,
                            const float* __restrict__ b1,
                            const float* __restrict__ W2,
                            const float* __restrict__ b2,
                            const float* __restrict__ aE,
                            const float* __restrict__ aC,
                            const float* __restrict__ aN,
                            float* __restrict__ out,
                            int Btot)
{
  extern __shared__ char smem[];
  _Float16* W1s   = (_Float16*)smem;                  // [256][WSTR]
  _Float16* W2s   = W1s + W_ELEMS;
  _Float16* tiles = W2s + W_ELEMS;                    // 4 waves x [16][WSTR]
  float*    ancs  = (float*)(tiles + 4 * TILE_ELEMS); // 3 x 256 normalized anchors
  float*    b1s   = ancs + 768;
  float*    b2s   = b1s + 256;
  float*    invn  = b2s + 256;                        // 3 (+1 pad)

  const int tid  = threadIdx.x;
  const int wave = tid >> 5;
  const int lane = tid & 31;
  const int ln   = lane & 15;
  const int hg   = lane >> 4;
  _Float16* tile = tiles + wave * TILE_ELEMS;

  // ---- stage weights (f32 -> f16, padded rows), biases, normalized anchors ----
  for (int i = tid; i < 256 * 256; i += 128) {
    int j = i >> 8, k = i & 255;
    W1s[j * WSTR + k] = (_Float16)W1[i];
    W2s[j * WSTR + k] = (_Float16)W2[i];
  }
  for (int i = tid; i < 256; i += 128) { b1s[i] = b1[i]; b2s[i] = b2[i]; }
  if (tid < 3) {
    const float* ap = (tid == 0) ? aE : (tid == 1) ? aC : aN;
    float s = 0.f;
    for (int c = 0; c < 256; ++c) { float v = ap[c]; s = fmaf(v, v, s); }
    invn[tid] = 1.0f / fmaxf(sqrtf(s), 1e-12f);
  }
  __syncthreads();
  for (int i = tid; i < 768; i += 128) {
    int a = i >> 8, c = i & 255;
    const float* ap = (a == 0) ? aE : (a == 1) ? aC : aN;
    ancs[i] = ap[c] * invn[a];
  }
  __syncthreads();

  // ---- load this wave's 16 rows of h into fp32 registers (WMMA C-layout) ----
  const int rowBase = blockIdx.x * 64 + wave * 16;
  float hreg[16][8];
  #pragma unroll
  for (int t = 0; t < 16; ++t)
    #pragma unroll
    for (int r = 0; r < 8; ++r)
      hreg[t][r] = h0[(size_t)(rowBase + r + 8 * hg) * DIM + t * 16 + ln];

  const size_t TOFF = (size_t)Btot * DIM;          // start of align trace
  const size_t QSZ  = (size_t)LAYERS * Btot * 3;   // per-quantity trace size

  for (int layer = 0; layer < LAYERS; ++layer) {
    // ---- per-row reductions: ||h||^2 and 3 anchor dot products ----
    float S[8], d0[8], d1[8], d2[8];
    #pragma unroll
    for (int r = 0; r < 8; ++r) { S[r] = 0.f; d0[r] = 0.f; d1[r] = 0.f; d2[r] = 0.f; }
    #pragma unroll
    for (int t = 0; t < 16; ++t) {
      int col = t * 16 + ln;
      float a0 = ancs[col], a1 = ancs[256 + col], a2 = ancs[512 + col];
      #pragma unroll
      for (int r = 0; r < 8; ++r) {
        float hv = hreg[t][r];
        S[r]  = fmaf(hv, hv, S[r]);
        d0[r] = fmaf(hv, a0, d0[r]);
        d1[r] = fmaf(hv, a1, d1[r]);
        d2[r] = fmaf(hv, a2, d2[r]);
      }
    }
    #pragma unroll
    for (int m = 1; m < 16; m <<= 1) {
      #pragma unroll
      for (int r = 0; r < 8; ++r) {
        S[r]  += __shfl_xor(S[r],  m, 32);
        d0[r] += __shfl_xor(d0[r], m, 32);
        d1[r] += __shfl_xor(d1[r], m, 32);
        d2[r] += __shfl_xor(d2[r], m, 32);
      }
    }
    float c0[8], c1[8], c2[8];
    #pragma unroll
    for (int r = 0; r < 8; ++r) {
      float nrm = sqrtf(S[r]);
      float ih  = 1.0f / fmaxf(nrm, 1e-12f);
      float al0 = d0[r] * ih, al1 = d1[r] * ih, al2 = d2[r] * ih;
      float dv0 = 1.f - al0, dv1 = 1.f - al1, dv2 = 1.f - al2;
      float dn0 = fmaxf(sqrtf(S[r] - 2.f * d0[r] + 1.f), 1e-12f);
      float dn1 = fmaxf(sqrtf(S[r] - 2.f * d1[r] + 1.f), 1e-12f);
      float dn2 = fmaxf(sqrtf(S[r] - 2.f * d2[r] + 1.f), 1e-12f);
      c0[r] = -0.1f * dv0 / dn0;
      c1[r] = -0.1f * dv1 / dn1;
      c2[r] = -0.1f * dv2 / dn2;
      if (ln < 3) {
        float al = (ln == 0) ? al0 : (ln == 1) ? al1 : al2;
        float dv = 1.f - al;
        size_t idx = ((size_t)layer * Btot + (rowBase + r + 8 * hg)) * 3 + ln;
        out[TOFF + idx]           = al;
        out[TOFF + QSZ + idx]     = dv;
        out[TOFF + 2 * QSZ + idx] = dv * dv;
      }
    }

    // ---- stage h as f16 in A-layout scratch ----
    #pragma unroll
    for (int t = 0; t < 16; ++t)
      #pragma unroll
      for (int r = 0; r < 8; ++r)
        tile[(r + 8 * hg) * WSTR + t * 16 + ln] = (_Float16)hreg[t][r];
    __syncthreads();

    // ---- GEMM1: t = tanh(h @ W1^T + b1) ----
    v16h afr[8];
    #pragma unroll
    for (int kk = 0; kk < 8; ++kk) afr[kk] = ldA(tile, ln, kk * 32, hg);
    #pragma unroll 2   // two independent acc chains -> deeper ds_load/wmma pipelining
    for (int t = 0; t < 16; ++t) {
      float bv = b1s[t * 16 + ln];
      v8f acc = { bv, bv, bv, bv, bv, bv, bv, bv };
      #pragma unroll
      for (int kk = 0; kk < 8; ++kk) {
        v16h bf = ldB(W1s, t * 16 + ln, kk * 32, hg);
        acc = __builtin_amdgcn_wmma_f32_16x16x32_f16(false, afr[kk], false, bf,
                                                     (short)0, acc, false, false);
      }
      #pragma unroll
      for (int r = 0; r < 8; ++r)
        tile[(r + 8 * hg) * WSTR + t * 16 + ln] = (_Float16)fast_tanh(acc[r]);
    }
    __syncthreads();

    // ---- GEMM2 + fused anchor-force / residual epilogue ----
    #pragma unroll
    for (int kk = 0; kk < 8; ++kk) afr[kk] = ldA(tile, ln, kk * 32, hg);
    float S2[8];
    #pragma unroll
    for (int r = 0; r < 8; ++r) S2[r] = 0.f;
    #pragma unroll 2   // two independent acc chains -> deeper ds_load/wmma pipelining
    for (int t = 0; t < 16; ++t) {
      float bv = b2s[t * 16 + ln];
      v8f acc = { bv, bv, bv, bv, bv, bv, bv, bv };
      #pragma unroll
      for (int kk = 0; kk < 8; ++kk) {
        v16h bf = ldB(W2s, t * 16 + ln, kk * 32, hg);
        acc = __builtin_amdgcn_wmma_f32_16x16x32_f16(false, afr[kk], false, bf,
                                                     (short)0, acc, false, false);
      }
      int col = t * 16 + ln;
      float a0 = ancs[col], a1 = ancs[256 + col], a2 = ancs[512 + col];
      #pragma unroll
      for (int r = 0; r < 8; ++r) {
        float hv = hreg[t][r];
        float hn = hv + acc[r]
                 + c0[r] * (hv - a0) + c1[r] * (hv - a1) + c2[r] * (hv - a2);
        hreg[t][r] = hn;
        S2[r] = fmaf(hn, hn, S2[r]);
      }
    }
    #pragma unroll
    for (int m = 1; m < 16; m <<= 1) {
      #pragma unroll
      for (int r = 0; r < 8; ++r) S2[r] += __shfl_xor(S2[r], m, 32);
    }
    #pragma unroll
    for (int r = 0; r < 8; ++r) {
      float n2 = sqrtf(S2[r]);
      float sc = (n2 > 10.f) ? (10.f / (n2 + 1e-8f)) : 1.f;
      #pragma unroll
      for (int t = 0; t < 16; ++t) hreg[t][r] *= sc;
    }
  }

  // ---- write h_final ----
  #pragma unroll
  for (int t = 0; t < 16; ++t)
    #pragma unroll
    for (int r = 0; r < 8; ++r)
      out[(size_t)(rowBase + r + 8 * hg) * DIM + t * 16 + ln] = hreg[t][r];
}

extern "C" void kernel_launch(void* const* d_in, const int* in_sizes, int n_in,
                              void* d_out, int out_size, void* d_ws, size_t ws_size,
                              hipStream_t stream) {
  const float* h0 = (const float*)d_in[0];
  const float* W1 = (const float*)d_in[1];
  const float* b1 = (const float*)d_in[2];
  const float* W2 = (const float*)d_in[3];
  const float* b2 = (const float*)d_in[4];
  const float* aE = (const float*)d_in[5];
  const float* aC = (const float*)d_in[6];
  const float* aN = (const float*)d_in[7];
  float* out = (float*)d_out;

  const int Btot = in_sizes[0] / DIM;     // 65536
  dim3 grid(Btot / 64);                   // 64 rows per workgroup (4 waves x 16)
  dim3 block(128);
  collapse_engine_kernel<<<grid, block, SMEM_BYTES, stream>>>(
      h0, W1, b1, W2, b2, aE, aC, aN, out, Btot);
}
